// ConvolutionalSelfAttention_3384434229344
// MI455X (gfx1250) — matrix operand order repaired and checked by reading, not compiled, not run on hardware
//
#include <hip/hip_runtime.h>

typedef __bf16 bf16;
typedef __attribute__((ext_vector_type(16))) __bf16 v16bf;
typedef __attribute__((ext_vector_type(8)))  float  v8f;

#define N_SP 3136      // 56*56
#define C_IN 256
#define KTOT 2304      // 256*9
#define NB   8         // batch

__device__ __forceinline__ bf16 f2bf(float f) {
    unsigned u = __builtin_bit_cast(unsigned, f);
    unsigned r = u + 0x7FFFu + ((u >> 16) & 1u);   // round-to-nearest-even
    unsigned short h = (unsigned short)(r >> 16);
    return __builtin_bit_cast(bf16, h);
}
__device__ __forceinline__ unsigned pack2(float lo, float hi) {
    unsigned a = (unsigned)__builtin_bit_cast(unsigned short, f2bf(lo));
    unsigned b = (unsigned)__builtin_bit_cast(unsigned short, f2bf(hi));
    return a | (b << 16);
}

// Fragment = 512 bf16 (1 KB), per-lane register order: lane's 16 bf16 at [lane*16..+15].
__device__ __forceinline__ v16bf read_frag_lds(const unsigned* p, int lane) {
    return ((const v16bf*)p)[lane];
}
__device__ __forceinline__ v16bf read_frag_g(const bf16* p, int lane) {
    return ((const v16bf*)p)[lane];
}

// A-fragment element address for matrix elem (row, k):
//   half = (k>>3)&1 ; e = (k&7) + 8*(k>>4) ; lane = 16*half + row
__device__ __forceinline__ int a_elem(int row, int k) {
    return ((((k >> 3) & 1) << 4) + row) * 16 + (k & 7) + ((k >> 4) << 3);
}
// B-fragment element address for matrix elem (k, col): half = k>>4 ; e = k&15
__device__ __forceinline__ int b_elem(int k, int col) {
    return (((k >> 4) << 4) + col) * 16 + (k & 15);
}

#define WMMA_BF16(A, B, C) \
    __builtin_amdgcn_wmma_f32_16x16x32_bf16(false, (A), false, (B), (short)0, (C), false, false)

// Workspace fragment indexing for attention operands:
//  Q (A-frag): frag (b, i>>4)                      elem a_elem(i&15, d)
//  K (B-frag): frag (b, j>>4)                      elem a_elem(j&15, d)
//  V (B-frag): frag ((b*98 + (j>>5))*16 + (c>>4))  elem: k=j within 32-step, col=c
__device__ __forceinline__ int frag_store_idx(int layout, int b, int m, int nl) {
    if (layout == 0) {        // Q
        return ((b * 196 + (nl >> 4)) << 9) + a_elem(nl & 15, m);
    } else if (layout == 2) { // K
        return ((b * 196 + (nl >> 4)) << 9) + a_elem(nl & 15, m);
    } else {                  // V
        return (((b * 98 + (nl >> 5)) * 16 + (m >> 4)) << 9)
               + ((((nl >> 3) & 1) << 4) + (m & 15)) * 16 + (nl & 7) + (((nl >> 4) & 1) << 3);
    }
}

// ---------------------------------------------------------------------------
// Prep 1: x f32 -> bf16 mirror (same layout), 4 elems/thread.
// ---------------------------------------------------------------------------
__global__ __launch_bounds__(256)
void cvt_x_kernel(const float* __restrict__ x, bf16* __restrict__ xb) {
    int i = (blockIdx.x * 256 + threadIdx.x) * 4;
    float4 v = *(const float4*)(x + i);
    uint2 p;
    p.x = pack2(v.x, v.y);
    p.y = pack2(v.z, v.w);
    *(uint2*)(xb + i) = p;
}

// ---------------------------------------------------------------------------
// Prep 2: weights f32 [Cout][Cin][3][3] -> bf16 A-fragment order with
// tap-major K: kk = (kh*3+kw)*256 + cin. Frag id = (m>>4)*72 + (kk>>5).
// ---------------------------------------------------------------------------
__global__ __launch_bounds__(256)
void swz_w_kernel(const float* __restrict__ w, bf16* __restrict__ wf, int total) {
    int idx = blockIdx.x * 256 + threadIdx.x;
    if (idx >= total) return;
    int m   = idx / KTOT;
    int r   = idx - m * KTOT;     // cin*9 + tap
    int cin = r / 9;
    int tap = r - cin * 9;
    int kk  = tap * 256 + cin;
    wf[(((m >> 4) * 72 + (kk >> 5)) << 9) + a_elem(m & 15, kk & 31)] = f2bf(w[idx]);
}

// ---------------------------------------------------------------------------
// Kernel 1: implicit-GEMM conv 3x3 SAME (tap-major K order).
// LDS-free, barrier-free: each lane gathers its own B-fragment slice straight
// from the bf16 x-mirror (lane 16h+c holds k = 8h+e (e<8), 16+8h+e-8 (e>=8) of spatial column c, a
// fixed stride-N_SP gather foldable into load immediate offsets). Weights are
// direct-global A fragments. Block tile (16*MT)M x 64N, 4 waves.
// ---------------------------------------------------------------------------
template<int MT>
__global__ __launch_bounds__(128)
void conv_qkv_kernel(const bf16* __restrict__ xb, const bf16* __restrict__ wf,
                     const float* __restrict__ bias, bf16* __restrict__ out,
                     int layout) {
    const int tid  = threadIdx.x;
    const int lane = tid & 31;
    const int wave = tid >> 5;
    const int n_base  = blockIdx.x * 64;        // 64 | 3136
    const int m_base  = blockIdx.y * (16 * MT);
    const int b       = n_base / N_SP;
    const int nl_base = n_base - b * N_SP;

    const int half = lane >> 4;
    const int col  = lane & 15;
    const int nl   = nl_base + wave * 16 + col;   // this lane's spatial column
    const int h    = nl / 56;
    const int wc   = nl - h * 56;

    v8f acc[MT];
#pragma unroll
    for (int mt = 0; mt < MT; ++mt) acc[mt] = (v8f){0.f,0.f,0.f,0.f,0.f,0.f,0.f,0.f};

    const bf16 BZ = __builtin_bit_cast(bf16, (unsigned short)0);
    const int xb_batch = b * (C_IN * N_SP);

    for (int tap = 0; tap < 9; ++tap) {
        int kh = tap / 3;
        int kw = tap - kh * 3;
        int ih = h + kh - 1;
        int iw = wc + kw - 1;
        bool ok = ((unsigned)ih < 56u) && ((unsigned)iw < 56u);
        // address of k = 8*half + 0 channel for this lane's pixel
        int a0 = xb_batch + (8 * half) * N_SP + ih * 56 + iw;

        for (int c8 = 0; c8 < 8; ++c8) {        // cin blocks of 32
            int ks = tap * 8 + c8;
            v16bf bfr;
#pragma unroll
            for (int e = 0; e < 16; ++e) bfr[e] = BZ;
            if (ok) {
#pragma unroll
                for (int e = 0; e < 16; ++e) bfr[e] = xb[a0 + (e + ((e >> 3) << 3)) * N_SP];
            }
            a0 += 32 * N_SP;
#pragma unroll
            for (int mt = 0; mt < MT; ++mt) {
                v16bf af = read_frag_g(wf + ((((m_base >> 4) + mt) * 72 + ks) << 9), lane);
                acc[mt] = WMMA_BF16(af, bfr, acc[mt]);
            }
        }
    }

    const int nl_out = nl;   // same column this lane computed
#pragma unroll
    for (int mt = 0; mt < MT; ++mt) {
#pragma unroll
        for (int r = 0; r < 8; ++r) {
            int m = m_base + mt * 16 + r + 8 * half;
            out[frag_store_idx(layout, b, m, nl_out)] = f2bf(acc[mt][r] + bias[m]);
        }
    }
}

// ---------------------------------------------------------------------------
// Kernel 2: softmax statistics (row max, sum exp). No LDS, no barriers.
// ---------------------------------------------------------------------------
__global__ __launch_bounds__(128)
void attn_stats_kernel(const bf16* __restrict__ q, const bf16* __restrict__ kx,
                       float* __restrict__ mws, float* __restrict__ sws) {
    const int tid  = threadIdx.x;
    const int lane = tid & 31;
    const int wave = tid >> 5;
    const int b    = blockIdx.y;
    const int i0   = blockIdx.x * 64 + wave * 16;
    const int half = lane >> 4;
    const int rl   = lane & 15;

    const v16bf aq = read_frag_g(q + ((b * 196 + (i0 >> 4)) << 9), lane);

    float vmax[8];
#pragma unroll
    for (int r = 0; r < 8; ++r) vmax[r] = -1e30f;

    for (int j0 = 0; j0 < N_SP; j0 += 64) {
#pragma unroll
        for (int t = 0; t < 4; ++t) {
            v16bf bk = read_frag_g(kx + ((b * 196 + (j0 >> 4) + t) << 9), lane);
            v8f e = {0.f,0.f,0.f,0.f,0.f,0.f,0.f,0.f};
            e = WMMA_BF16(aq, bk, e);
#pragma unroll
            for (int r = 0; r < 8; ++r) vmax[r] = fmaxf(vmax[r], e[r]);
        }
    }
#pragma unroll
    for (int msk = 8; msk >= 1; msk >>= 1)
#pragma unroll
        for (int r = 0; r < 8; ++r)
            vmax[r] = fmaxf(vmax[r], __shfl_xor(vmax[r], msk, 32));

    float vsum[8];
#pragma unroll
    for (int r = 0; r < 8; ++r) vsum[r] = 0.f;

    for (int j0 = 0; j0 < N_SP; j0 += 64) {
#pragma unroll
        for (int t = 0; t < 4; ++t) {
            v16bf bk = read_frag_g(kx + ((b * 196 + (j0 >> 4) + t) << 9), lane);
            v8f e = {0.f,0.f,0.f,0.f,0.f,0.f,0.f,0.f};
            e = WMMA_BF16(aq, bk, e);
#pragma unroll
            for (int r = 0; r < 8; ++r) vsum[r] += __expf(e[r] - vmax[r]);
        }
    }
#pragma unroll
    for (int msk = 8; msk >= 1; msk >>= 1)
#pragma unroll
        for (int r = 0; r < 8; ++r)
            vsum[r] += __shfl_xor(vsum[r], msk, 32);

    if (rl == 0) {
#pragma unroll
        for (int r = 0; r < 8; ++r) {
            int ig = i0 + 8 * half + r;
            mws[b * N_SP + ig] = vmax[r];
            sws[b * N_SP + ig] = vsum[r];
        }
    }
}

// ---------------------------------------------------------------------------
// Kernel 3: streaming output, fully independent waves (no __syncthreads).
// ---------------------------------------------------------------------------
__global__ __launch_bounds__(128)
void attn_out_kernel(const bf16* __restrict__ q, const bf16* __restrict__ kx,
                     const bf16* __restrict__ vt,
                     const float* __restrict__ mws, const float* __restrict__ sws,
                     const float* __restrict__ x, const float* __restrict__ gamma_p,
                     float* __restrict__ out) {
    __shared__ __align__(32) unsigned Pfrag[4][256];  // private per wave

    const int tid  = threadIdx.x;
    const int lane = tid & 31;
    const int wave = tid >> 5;
    const int b    = blockIdx.y;
    const int i0   = blockIdx.x * 32;
    const int it   = wave >> 1;
    const int cs   = wave & 1;
    const int half = lane >> 4;
    const int rl   = lane & 15;
    const int itile = (i0 >> 4) + it;

    const v16bf aq = read_frag_g(q + ((b * 196 + itile) << 9), lane);

    float mrow[8];
#pragma unroll
    for (int r = 0; r < 8; ++r)
        mrow[r] = mws[b * N_SP + itile * 16 + 8 * half + r];

    v8f acc[8];
#pragma unroll
    for (int ct = 0; ct < 8; ++ct)
        acc[ct] = (v8f){0.f,0.f,0.f,0.f,0.f,0.f,0.f,0.f};

    bf16* pb = (bf16*)&Pfrag[wave][0];
    const int ha = (rl >> 3) & 1;
    for (int j0 = 0; j0 < N_SP; j0 += 32) {
        v16bf bk0 = read_frag_g(kx + ((b * 196 + (j0 >> 4))     << 9), lane);
        v16bf bk1 = read_frag_g(kx + ((b * 196 + (j0 >> 4) + 1) << 9), lane);
        v8f e0 = {0.f,0.f,0.f,0.f,0.f,0.f,0.f,0.f};
        v8f e1 = {0.f,0.f,0.f,0.f,0.f,0.f,0.f,0.f};
        e0 = WMMA_BF16(aq, bk0, e0);
        e1 = WMMA_BF16(aq, bk1, e1);

        // transpose D->A via private LDS region (same-wave write->read)
#pragma unroll
        for (int r = 0; r < 8; ++r) {
            int row  = r + 8 * half;
            int base = ((ha << 4) + row) * 16;
            pb[base + (rl & 7)]     = f2bf(__expf(e0[r] - mrow[r]));   // j = rl
            pb[base + (rl & 7) + 8] = f2bf(__expf(e1[r] - mrow[r]));   // j = rl+16
        }
        v16bf pf = read_frag_lds(Pfrag[wave], lane);

        const bf16* vbase = vt + (((b * 98 + (j0 >> 5)) * 16 + cs * 8) << 9);
#pragma unroll
        for (int ct = 0; ct < 8; ++ct) {
            v16bf bv = read_frag_g(vbase + (ct << 9), lane);
            acc[ct] = WMMA_BF16(pf, bv, acc[ct]);
        }
    }

    const float gamma = gamma_p[0];
    float inv_s[8];
#pragma unroll
    for (int r = 0; r < 8; ++r)
        inv_s[r] = 1.f / sws[b * N_SP + itile * 16 + 8 * half + r];

#pragma unroll
    for (int ct = 0; ct < 8; ++ct) {
#pragma unroll
        for (int r = 0; r < 8; ++r) {
            int c  = cs * 128 + ct * 16 + rl;
            int ig = itile * 16 + 8 * half + r;
            int idx = (b * C_IN + c) * N_SP + ig;
            out[idx] = gamma * acc[ct][r] * inv_s[r] + x[idx];
        }
    }
}

// ---------------------------------------------------------------------------
extern "C" void kernel_launch(void* const* d_in, const int* in_sizes, int n_in,
                              void* d_out, int out_size, void* d_ws, size_t ws_size,
                              hipStream_t stream) {
    (void)in_sizes; (void)n_in; (void)out_size; (void)ws_size;
    const float* x     = (const float*)d_in[0];
    const float* wq    = (const float*)d_in[1];
    const float* bq    = (const float*)d_in[2];
    const float* wk    = (const float*)d_in[3];
    const float* bk    = (const float*)d_in[4];
    const float* wv    = (const float*)d_in[5];
    const float* bv    = (const float*)d_in[6];
    const float* gamma = (const float*)d_in[7];
    float* out = (float*)d_out;

    const int qk_elems = NB * 32 * N_SP;      // 802816
    const int v_elems  = NB * N_SP * C_IN;    // 6422528
    const int x_elems  = NB * C_IN * N_SP;    // 6422528
    const int wqk_el   = 32 * KTOT;           // 73728
    const int wv_el    = 256 * KTOT;          // 589824

    bf16* qws = (bf16*)d_ws;
    bf16* kws = qws + qk_elems;
    bf16* vws = kws + qk_elems;
    bf16* xbw = vws + v_elems;
    bf16* wqf = xbw + x_elems;
    bf16* wkf = wqf + wqk_el;
    bf16* wvf = wkf + wqk_el;
    float* mws = (float*)(wvf + wv_el);
    float* sws = mws + NB * N_SP;

    dim3 blk256(256);
    cvt_x_kernel<<<x_elems / (256 * 4), blk256, 0, stream>>>(x, xbw);
    swz_w_kernel<<<(wqk_el + 255) / 256, blk256, 0, stream>>>(wq, wqf, wqk_el);
    swz_w_kernel<<<(wqk_el + 255) / 256, blk256, 0, stream>>>(wk, wkf, wqk_el);
    swz_w_kernel<<<(wv_el  + 255) / 256, blk256, 0, stream>>>(wv, wvf, wv_el);

    dim3 blk(128);
    dim3 g_qk(392, 1);   // 25088/64 spatial tiles, 32 channels (MT=2)
    dim3 g_v(392, 4);    // 256 channels, 64 per block (MT=4)
    conv_qkv_kernel<2><<<g_qk, blk, 0, stream>>>(xbw, wqf, bq, qws, 0);
    conv_qkv_kernel<2><<<g_qk, blk, 0, stream>>>(xbw, wkf, bk, kws, 2);
    conv_qkv_kernel<4><<<g_v,  blk, 0, stream>>>(xbw, wvf, bv, vws, 1);

    dim3 g_stats(49, NB);  // 3136/64 row-blocks per batch
    attn_stats_kernel<<<g_stats, blk, 0, stream>>>(qws, kws, mws, sws);

    dim3 g_out(98, NB);    // 3136/32 row-blocks per batch
    attn_out_kernel<<<g_out, blk, 0, stream>>>(qws, kws, vws, mws, sws, x, gamma, out);
}
